// HeCo_17025250361909
// MI455X (gfx1250) — matrix operand order repaired and checked by reading, not compiled, not run on hardware
//
#include <hip/hip_runtime.h>
#include <hip/hip_bf16.h>
#include <math.h>

typedef __attribute__((ext_vector_type(16))) _Float16 v16h;
typedef __attribute__((ext_vector_type(8)))  _Float16 v8h;
typedef __attribute__((ext_vector_type(8)))  float    v8f;

#define N_USERS 8192
#define N_TY1   16384
#define N_TY2   4096
#define DIN     128
#define DH      64
#define NEDGE   131072
#define INV_TAU 1.25f   // 1/0.8

static __device__ __forceinline__ float elu1(float x)   { return x > 0.f ? x : (expf(x) - 1.f); }
static __device__ __forceinline__ float lrelu(float x)  { return x >= 0.f ? x : 0.01f * x; }

// ---------------------------------------------------------------- utilities
__global__ void fill_kernel(float* __restrict__ p, float v, int n) {
    int i = blockIdx.x * blockDim.x + threadIdx.x;
    if (i < n) p[i] = v;
}

// out[n,d] = act( sum_k in[n,k] * W[k,d] + b[d] ), 64 threads per row, runtime K<=128
__global__ void rowmm_kernel(const float* __restrict__ in, const float* __restrict__ W,
                             const float* __restrict__ b, float* __restrict__ out,
                             int K, int act /*0=none,1=elu*/) {
    __shared__ float xs[128];
    int n = blockIdx.x, d = threadIdx.x;
    for (int k = d; k < K; k += 64) xs[k] = in[(size_t)n * K + k];
    __syncthreads();
    float acc = b ? b[d] : 0.f;
    for (int k = 0; k < K; ++k) acc += xs[k] * W[k * DH + d];
    if (act == 1) acc = elu1(acc);
    out[(size_t)n * DH + d] = acc;
}

// sp[d] += tanh( (in[n,:] @ W + b)[d] ) / N   (semantic attention summary vector)
__global__ void tanh_mean_kernel(const float* __restrict__ in, const float* __restrict__ W,
                                 const float* __restrict__ b, float* __restrict__ sp, int N) {
    __shared__ float xs[64];
    int n = blockIdx.x, d = threadIdx.x;
    xs[d] = in[(size_t)n * DH + d];
    __syncthreads();
    float acc = b[d];
    for (int k = 0; k < DH; ++k) acc += xs[k] * W[k * DH + d];
    atomicAdd(&sp[d], tanhf(acc) * (1.f / (float)N));
}

// beta = softmax([sp0 . attv, sp1 . attv])
__global__ void beta_kernel(const float* __restrict__ sp, const float* __restrict__ attv,
                            float* __restrict__ beta) {
    __shared__ float r0[64], r1[64];
    int d = threadIdx.x;
    r0[d] = sp[d] * attv[d];
    r1[d] = sp[64 + d] * attv[d];
    __syncthreads();
    for (int off = 32; off >= 1; off >>= 1) {
        if (d < off) { r0[d] += r0[d + off]; r1[d] += r1[d + off]; }
        __syncthreads();
    }
    if (d == 0) {
        float a = r0[0], c = r1[0];
        float mx = fmaxf(a, c);
        float ea = expf(a - mx), ec = expf(c - mx);
        beta[0] = ea / (ea + ec);
        beta[1] = ec / (ea + ec);
    }
}

__global__ void blend_kernel(const float* __restrict__ e0, const float* __restrict__ e1,
                             const float* __restrict__ beta, float* __restrict__ z, int n) {
    int i = blockIdx.x * blockDim.x + threadIdx.x;
    if (i < n) z[i] = beta[0] * e0[i] + beta[1] * e1[i];
}

// ---------------------------------------------------------------- GCN
__global__ void deg_kernel(const int* __restrict__ dst, float* __restrict__ deg, int E) {
    int e = blockIdx.x * blockDim.x + threadIdx.x;
    if (e < E) atomicAdd(&deg[dst[e]], 1.f);
}

__global__ void gcn_scatter_kernel(const float* __restrict__ hw, const int* __restrict__ src,
                                   const int* __restrict__ dst, const float* __restrict__ deg,
                                   float* __restrict__ agg, int E) {
    int idx = blockIdx.x * blockDim.x + threadIdx.x;
    if (idx >= E * DH) return;
    int e = idx >> 6, d = idx & 63;
    int s = src[e], t = dst[e];
    float nrm = rsqrtf(deg[s]) * rsqrtf(deg[t]);
    atomicAdd(&agg[(size_t)t * DH + d], hw[(size_t)s * DH + d] * nrm);
}

__global__ void gcn_finalize_kernel(float* __restrict__ agg, const float* __restrict__ hw,
                                    const float* __restrict__ deg, const float* __restrict__ b,
                                    const float* __restrict__ a_p, int N) {
    int idx = blockIdx.x * blockDim.x + threadIdx.x;
    if (idx >= N * DH) return;
    int n = idx >> 6, d = idx & 63;
    float v = agg[idx] + hw[idx] / deg[n] + b[d];
    agg[idx] = v >= 0.f ? v : a_p[0] * v;   // PReLU
}

// ---------------------------------------------------------------- intra (schema) attention
__global__ void intra_kernel(const int* __restrict__ nei, int S,
                             const float* __restrict__ ht, const float* __restrict__ hu,
                             const float* __restrict__ attv,  // [0:64]=ref part, [64:128]=nei part
                             float* __restrict__ out) {
    __shared__ float nes[32 * 64];
    __shared__ float red[64];
    __shared__ float logits[32];
    __shared__ float cref;
    int n = blockIdx.x, d = threadIdx.x;
    red[d] = hu[(size_t)n * DH + d] * attv[d];
    for (int s = 0; s < S; ++s) {
        int idx = nei[(size_t)n * S + s];
        nes[s * DH + d] = ht[(size_t)idx * DH + d];
    }
    __syncthreads();
    for (int off = 32; off >= 1; off >>= 1) {
        if (d < off) red[d] += red[d + off];
        __syncthreads();
    }
    if (d == 0) cref = red[0];
    __syncthreads();
    if (d < S) {
        float acc = 0.f;
        for (int k = 0; k < DH; ++k) acc += nes[d * DH + k] * attv[64 + k];
        logits[d] = lrelu(cref + acc);
    }
    __syncthreads();
    float mx = -1e30f;
    for (int s = 0; s < S; ++s) mx = fmaxf(mx, logits[s]);
    float den = 0.f, acc = 0.f;
    for (int s = 0; s < S; ++s) {
        float w = expf(logits[s] - mx);
        den += w;
        acc += w * nes[s * DH + d];
    }
    out[(size_t)n * DH + d] = elu1(acc / den);
}

// ---------------------------------------------------------------- normalize + pack f16
__global__ void norm_f16_kernel(const float* __restrict__ z, _Float16* __restrict__ zn) {
    __shared__ float red[64];
    __shared__ float sc;
    int n = blockIdx.x, d = threadIdx.x;
    float v = z[(size_t)n * DH + d];
    red[d] = v * v;
    __syncthreads();
    for (int off = 32; off >= 1; off >>= 1) {
        if (d < off) red[d] += red[d + off];
        __syncthreads();
    }
    if (d == 0) sc = rsqrtf(red[0]);
    __syncthreads();
    zn[(size_t)n * DH + d] = (_Float16)(v * sc);
}

// ---------------------------------------------------------------- fused cosine-sim / exp / reductions (WMMA)
// A frag (16x32 f16): per lane, K blocks {0..7,16..23} (lanes 0-15) / {8..15,24..31} (lanes 16-31)
static __device__ __forceinline__ v16h load_fragA(const _Float16* __restrict__ z, int row, int kc, int hi16) {
    const _Float16* p = z + (size_t)row * DH + kc * 32 + hi16 * 8;
    v8h lo = *(const v8h*)p;
    v8h hi = *(const v8h*)(p + 16);
    v16h a;
#pragma unroll
    for (int i = 0; i < 8; ++i) { a[i] = lo[i]; a[i + 8] = hi[i]; }
    return a;
}
// B frag (32x16 f16): per lane (column = lane&15), K blocks {0..7,16..23} (lanes 0-15) / {8..15,24..31} (lanes 16-31)
static __device__ __forceinline__ v16h load_fragB(const _Float16* __restrict__ z, int row, int kc, int hi16) {
    const _Float16* p = z + (size_t)row * DH + kc * 32 + hi16 * 8;
    v8h lo = *(const v8h*)p;
    v8h hi = *(const v8h*)(p + 16);
    v16h b;
#pragma unroll
    for (int i = 0; i < 8; ++i) { b[i] = lo[i]; b[i + 8] = hi[i]; }
    return b;
}

__global__ __launch_bounds__(256)
void sim_wmma_kernel(const _Float16* __restrict__ znA, const _Float16* __restrict__ znB,
                     const float* __restrict__ pos,
                     float* __restrict__ rowsum, float* __restrict__ colsum,
                     float* __restrict__ posr, float* __restrict__ posc) {
    __shared__ float lds_cs[128];
    __shared__ float lds_pc[128];
    int tid = threadIdx.x;
    if (tid < 128) { lds_cs[tid] = 0.f; lds_pc[tid] = 0.f; }
    __syncthreads();

    int lane = tid & 31, w = tid >> 5;
    int l15 = lane & 15, hi16 = lane >> 4;
    int Ib = blockIdx.y * 128, Jb = blockIdx.x * 128;
    int gi = Ib + w * 16;             // this wave's 16-row strip

    v16h a0 = load_fragA(znA, gi + l15, 0, hi16);
    v16h a1 = load_fragA(znA, gi + l15, 1, hi16);

    float racc[8], pracc[8];
#pragma unroll
    for (int r = 0; r < 8; ++r) { racc[r] = 0.f; pracc[r] = 0.f; }

    for (int jt = 0; jt < 8; ++jt) {
        int gj = Jb + jt * 16;
        v16h b0 = load_fragB(znB, gj + l15, 0, hi16);
        v16h b1 = load_fragB(znB, gj + l15, 1, hi16);
        v8f c = {0.f, 0.f, 0.f, 0.f, 0.f, 0.f, 0.f, 0.f};
        c = __builtin_amdgcn_wmma_f32_16x16x32_f16(false, a0, false, b0, (short)0, c, false, false);
        c = __builtin_amdgcn_wmma_f32_16x16x32_f16(false, a1, false, b1, (short)0, c, false, false);

        // transposed pos tile: row gj+l15, 8 contiguous cols starting gi + hi16*8
        const float* pt = pos + (size_t)(gj + l15) * N_USERS + gi + hi16 * 8;
        float4 t0 = *(const float4*)pt;
        float4 t1 = *(const float4*)(pt + 4);
        float post[8] = {t0.x, t0.y, t0.z, t0.w, t1.x, t1.y, t1.z, t1.w};

        float s = 0.f, pc = 0.f;
#pragma unroll
        for (int r = 0; r < 8; ++r) {
            float mval = expf(c[r] * INV_TAU);   // exp(cos/tau); rows normalized upstream
            int M = r + hi16 * 8;
            float pd = pos[(size_t)(gi + M) * N_USERS + gj + l15];
            racc[r]  += mval;
            pracc[r] += mval * pd;
            s  += mval;
            pc += mval * post[r];
        }
        s  += __shfl_xor(s, 16);
        pc += __shfl_xor(pc, 16);
        if (lane < 16) {
            atomicAdd(&lds_cs[jt * 16 + l15], s);
            atomicAdd(&lds_pc[jt * 16 + l15], pc);
        }
    }

    // row reductions: sum over the 16 lanes of each half-wave
#pragma unroll
    for (int r = 0; r < 8; ++r) {
        float v = racc[r], pv = pracc[r];
        for (int off = 8; off >= 1; off >>= 1) {
            v  += __shfl_xor(v, off);
            pv += __shfl_xor(pv, off);
        }
        if (l15 == r) {
            int row = gi + r + hi16 * 8;
            atomicAdd(&rowsum[row], v);
            atomicAdd(&posr[row], pv);
        }
    }
    __syncthreads();
    if (tid < 128) {
        atomicAdd(&colsum[Jb + tid], lds_cs[tid]);
        atomicAdd(&posc[Jb + tid], lds_pc[tid]);
    }
}

// ---------------------------------------------------------------- final loss
__global__ void loss_kernel(const float* __restrict__ rowsum, const float* __restrict__ colsum,
                            const float* __restrict__ posr, const float* __restrict__ posc,
                            float* __restrict__ out) {
    __shared__ float red[256];
    float acc = 0.f;
    for (int i = threadIdx.x; i < N_USERS; i += 256) {
        float pr = posr[i] / (rowsum[i] + 1e-8f);
        float pc = posc[i] / (colsum[i] + 1e-8f);
        acc += 0.5f * logf(pr) + 0.5f * logf(pc);
    }
    red[threadIdx.x] = acc;
    __syncthreads();
    for (int off = 128; off >= 1; off >>= 1) {
        if (threadIdx.x < off) red[threadIdx.x] += red[threadIdx.x + off];
        __syncthreads();
    }
    if (threadIdx.x == 0) out[0] = -red[0] / (float)N_USERS;
}

// ---------------------------------------------------------------- host orchestration
extern "C" void kernel_launch(void* const* d_in, const int* in_sizes, int n_in,
                              void* d_out, int out_size, void* d_ws, size_t ws_size,
                              hipStream_t stream) {
    (void)in_sizes; (void)n_in; (void)out_size; (void)ws_size;
    const float* x_user   = (const float*)d_in[0];
    const float* x_t1     = (const float*)d_in[1];
    const float* x_t2     = (const float*)d_in[2];
    const float* pos      = (const float*)d_in[3];
    const float* map_W    = (const float*)d_in[4];
    const float* map_b    = (const float*)d_in[5];
    const float* gcn_W    = (const float*)d_in[6];
    const float* gcn_b    = (const float*)d_in[7];
    const float* gcn_a    = (const float*)d_in[8];
    const float* mp_fc_W  = (const float*)d_in[9];
    const float* mp_fc_b  = (const float*)d_in[10];
    const float* mp_att   = (const float*)d_in[11];
    const float* intra_att= (const float*)d_in[12];
    const float* sc_fc_W  = (const float*)d_in[13];
    const float* sc_fc_b  = (const float*)d_in[14];
    const float* sc_att   = (const float*)d_in[15];
    const float* proj_W1  = (const float*)d_in[16];
    const float* proj_b1  = (const float*)d_in[17];
    const float* proj_W2  = (const float*)d_in[18];
    const float* proj_b2  = (const float*)d_in[19];
    const int*   mp_edges = (const int*)d_in[20];
    const int*   nei0     = (const int*)d_in[21];
    const int*   nei1     = (const int*)d_in[22];

    float* ws = (float*)d_ws;
    size_t o = 0;
    float* h_user = ws + o; o += (size_t)N_USERS * DH;
    float* h_t1   = ws + o; o += (size_t)N_TY1 * DH;
    float* h_t2   = ws + o; o += (size_t)N_TY2 * DH;
    float* hw     = ws + o; o += (size_t)N_USERS * DH;
    float* emb0   = ws + o; o += (size_t)N_USERS * DH;
    float* emb1   = ws + o; o += (size_t)N_USERS * DH;
    float* z_mp   = ws + o; o += (size_t)N_USERS * DH;
    float* e0     = ws + o; o += (size_t)N_USERS * DH;
    float* e1     = ws + o; o += (size_t)N_USERS * DH;
    float* z_sc   = ws + o; o += (size_t)N_USERS * DH;
    float* tmp    = ws + o; o += (size_t)N_USERS * DH;
    float* zp_mp  = ws + o; o += (size_t)N_USERS * DH;
    float* zp_sc  = ws + o; o += (size_t)N_USERS * DH;
    float* deg    = ws + o; o += N_USERS;
    float* sp     = ws + o; o += 128;
    float* beta   = ws + o; o += 8;
    float* rowsum = ws + o; o += N_USERS;
    float* colsum = ws + o; o += N_USERS;
    float* posr   = ws + o; o += N_USERS;
    float* posc   = ws + o; o += N_USERS;
    _Float16* zn_mp = (_Float16*)(ws + o); o += (size_t)N_USERS * DH / 2;
    _Float16* zn_sc = (_Float16*)(ws + o); o += (size_t)N_USERS * DH / 2;

    const int NU64 = N_USERS * DH;

    // 1) per-type feature mapping + ELU
    rowmm_kernel<<<N_USERS, 64, 0, stream>>>(x_user, map_W,              map_b,       h_user, DIN, 1);
    rowmm_kernel<<<N_TY1,   64, 0, stream>>>(x_t1,   map_W + DIN * DH,   map_b + DH,  h_t1,   DIN, 1);
    rowmm_kernel<<<N_TY2,   64, 0, stream>>>(x_t2,   map_W + 2*DIN*DH,   map_b + 2*DH,h_t2,   DIN, 1);

    // 2) meta-path GCNs (P=2)
    for (int p = 0; p < 2; ++p) {
        float* emb = p ? emb1 : emb0;
        const int* src = mp_edges + (size_t)p * 2 * NEDGE;
        const int* dst = src + NEDGE;
        rowmm_kernel<<<N_USERS, 64, 0, stream>>>(h_user, gcn_W + p * DH * DH, nullptr, hw, DH, 0);
        fill_kernel<<<(N_USERS + 255) / 256, 256, 0, stream>>>(deg, 1.f, N_USERS);
        fill_kernel<<<(NU64 + 255) / 256, 256, 0, stream>>>(emb, 0.f, NU64);
        deg_kernel<<<(NEDGE + 255) / 256, 256, 0, stream>>>(dst, deg, NEDGE);
        gcn_scatter_kernel<<<(NEDGE * DH + 255) / 256, 256, 0, stream>>>(hw, src, dst, deg, emb, NEDGE);
        gcn_finalize_kernel<<<(NU64 + 255) / 256, 256, 0, stream>>>(emb, hw, deg, gcn_b + p * DH, gcn_a + p, N_USERS);
    }

    // 3) meta-path semantic attention fusion -> z_mp
    fill_kernel<<<1, 128, 0, stream>>>(sp, 0.f, 128);
    tanh_mean_kernel<<<N_USERS, 64, 0, stream>>>(emb0, mp_fc_W, mp_fc_b, sp,      N_USERS);
    tanh_mean_kernel<<<N_USERS, 64, 0, stream>>>(emb1, mp_fc_W, mp_fc_b, sp + 64, N_USERS);
    beta_kernel<<<1, 64, 0, stream>>>(sp, mp_att, beta);
    blend_kernel<<<(NU64 + 255) / 256, 256, 0, stream>>>(emb0, emb1, beta, z_mp, NU64);

    // 4) schema encoder: intra attention + fusion -> z_sc
    intra_kernel<<<N_USERS, 64, 0, stream>>>(nei0, 16, h_t1, h_user, intra_att,        e0);
    intra_kernel<<<N_USERS, 64, 0, stream>>>(nei1, 32, h_t2, h_user, intra_att + 128,  e1);
    fill_kernel<<<1, 128, 0, stream>>>(sp, 0.f, 128);
    tanh_mean_kernel<<<N_USERS, 64, 0, stream>>>(e0, sc_fc_W, sc_fc_b, sp,      N_USERS);
    tanh_mean_kernel<<<N_USERS, 64, 0, stream>>>(e1, sc_fc_W, sc_fc_b, sp + 64, N_USERS);
    beta_kernel<<<1, 64, 0, stream>>>(sp, sc_att, beta);
    blend_kernel<<<(NU64 + 255) / 256, 256, 0, stream>>>(e0, e1, beta, z_sc, NU64);

    // 5) projection MLP
    rowmm_kernel<<<N_USERS, 64, 0, stream>>>(z_mp, proj_W1, proj_b1, tmp,   DH, 1);
    rowmm_kernel<<<N_USERS, 64, 0, stream>>>(tmp,  proj_W2, proj_b2, zp_mp, DH, 0);
    rowmm_kernel<<<N_USERS, 64, 0, stream>>>(z_sc, proj_W1, proj_b1, tmp,   DH, 1);
    rowmm_kernel<<<N_USERS, 64, 0, stream>>>(tmp,  proj_W2, proj_b2, zp_sc, DH, 0);

    // 6) row-normalize + pack to f16 (cosine numerator folded into WMMA inputs)
    norm_f16_kernel<<<N_USERS, 64, 0, stream>>>(zp_mp, zn_mp);
    norm_f16_kernel<<<N_USERS, 64, 0, stream>>>(zp_sc, zn_sc);

    // 7) fused similarity matrix + reductions (rowsum/colsum/posr/posc are contiguous)
    fill_kernel<<<(4 * N_USERS + 255) / 256, 256, 0, stream>>>(rowsum, 0.f, 4 * N_USERS);
    dim3 g(N_USERS / 128, N_USERS / 128);
    sim_wmma_kernel<<<g, 256, 0, stream>>>(zn_mp, zn_sc, pos, rowsum, colsum, posr, posc);

    // 8) scalar loss
    loss_kernel<<<1, 256, 0, stream>>>(rowsum, colsum, posr, posc, (float*)d_out);
}
